// CTRGC_sparse_32006096290315
// MI455X (gfx1250) — matrix operand order repaired and checked by reading, not compiled, not run on hardware
//
#include <hip/hip_runtime.h>

// CTRGC: out = attn ⊗ (w3·x), attn = (w4·tanh(outerdiff(w1·xm, w2·xm)) + b4)*alpha + A, xm = mean_T(x)
// bf16 WMMA (fp32 accum) for the three GEMMs; T-mean folded into the GEMM1 kernel (x read once).
// x loads and out stores are non-temporal so the 100MB of intermediates (x3p+attnT) stay in the
// 192MB L2 across kernels -> effective HBM traffic ~277MB ~ 12us @ 23.3TB/s; WMMA compute ~4us.

#define NN   64
#define CIN  256
#define COUT 256
#define CR   32
#define TT   64
#define VV   25
#define U32  32
#define NPAD 640   // 625 (u,v) columns padded to 640

typedef __attribute__((ext_vector_type(16))) __bf16 v16bf;
typedef __attribute__((ext_vector_type(8)))  __bf16 v8bf;
typedef __attribute__((ext_vector_type(8)))  float  v8f;

union BF16x16 { v16bf v; v8bf h[2]; };
union V8F     { v8f v; float f[8]; };

// ---------------- K3: x3p[n][c][t][u32] = bf16(w3·x + b3)  (+ fold T-mean partials) ----------------
__global__ __launch_bounds__(512) void ctrgc_gemm1_mean_kernel(
    const float* __restrict__ x, const float* __restrict__ w3,
    const float* __restrict__ b3, __bf16* __restrict__ x3p,
    float* __restrict__ xm_part)          // [2][NN][CIN][VV]
{
    int bid = blockIdx.x;                 // n*16 + cb*2 + tb
    int tb = bid & 1, cb = (bid >> 1) & 7, n = bid >> 4;
    int c0 = cb * 32, t0 = tb * 32;

    __shared__ __attribute__((aligned(16))) __bf16 xs[32 * 32 * 32];  // [t][u32][ci32] 64KB
    __shared__ __attribute__((aligned(16))) __bf16 w3s[32 * 32];      // [c][ci32]

    int tid = threadIdx.x;
    int wave = tid >> 5, lane = tid & 31;
    int l16 = lane & 15; bool hi = lane >= 16;
    int ak = hi ? 8 : 0, bk = hi ? 8 : 0;

    // zero u-pad slots once (never touched by staging)
    for (int i = tid; i < 32 * 7 * 32; i += 512) {
        int t = i / (7 * 32); int rem = i % (7 * 32);
        int u = VV + rem / 32; int ci = rem & 31;
        xs[(t * 32 + u) * 32 + ci] = (__bf16)0.f;
    }

    V8F acc[2][2][2] = {};                // [tq][uhalf][m]
    const float* xn = x + (size_t)n * CIN * TT * VV;

    for (int kk = 0; kk < 8; ++kk) {
        int ci0 = kk * 32;
        __syncthreads();
        for (int i = tid; i < 1024; i += 512) {
            int c = i >> 5, ci = i & 31;
            w3s[i] = (__bf16)w3[(size_t)(c0 + c) * CIN + ci0 + ci];
        }
        // stage x chunk (streamed once -> non-temporal: keep L2 for x3p/attnT)
        for (int i = tid; i < 32 * 32 * 25; i += 512) {
            int ci = i / 800; int rem = i % 800;
            int t = rem / 25, v = rem % 25;
            float f = __builtin_nontemporal_load(
                xn + ((size_t)(ci0 + ci) * TT + t0 + t) * VV + v);
            xs[(t * 32 + v) * 32 + ci] = (__bf16)f;
        }

        // fold: deterministic T-mean partial sums (only c-block 0; data is L0/L2-hot)
        if (c0 == 0) {
            for (int col = tid; col < 32 * VV; col += 512) {
                int ci = col / VV, v = col % VV;
                const float* pr = xn + ((size_t)(ci0 + ci) * TT + t0) * VV + v;
                float s = 0.f;
                #pragma unroll 4
                for (int t = 0; t < 32; ++t) s += __builtin_nontemporal_load(pr + t * VV);
                xm_part[(((size_t)tb * NN + n) * CIN + ci0 + ci) * VV + v] = s;
            }
        }

        // prefetch next ci-chunk while this one is consumed by WMMA
        if (kk < 7) {
            for (int i = tid; i < 32 * 32; i += 512) {
                int ci = i >> 5, t = i & 31;
                __builtin_prefetch(xn + ((size_t)(ci0 + 32 + ci) * TT + t0 + t) * VV, 0, 1);
            }
        }
        __syncthreads();

        BF16x16 a[2];
        #pragma unroll
        for (int m = 0; m < 2; ++m) {
            int arow = m * 16 + l16;
            a[m].h[0] = *(const v8bf*)&w3s[arow * 32 + ak];
            a[m].h[1] = *(const v8bf*)&w3s[arow * 32 + ak + 16];
        }
        #pragma unroll
        for (int tq = 0; tq < 2; ++tq) {
            int t = wave * 2 + tq;
            #pragma unroll
            for (int uh = 0; uh < 2; ++uh) {
                BF16x16 b;
                int brow = t * 32 + uh * 16 + l16;
                b.h[0] = *(const v8bf*)&xs[brow * 32 + bk];
                b.h[1] = *(const v8bf*)&xs[brow * 32 + bk + 16];
                #pragma unroll
                for (int m = 0; m < 2; ++m)
                    acc[tq][uh][m].v = __builtin_amdgcn_wmma_f32_16x16x32_bf16(
                        false, a[m].v, false, b.v, (short)0, acc[tq][uh][m].v, false, false);
            }
        }
    }

    // epilogue: +b3, bf16, zero for u>=25 (exact-zero K-padding for GEMM2)
    #pragma unroll
    for (int tq = 0; tq < 2; ++tq) {
        int t = t0 + wave * 2 + tq;
        #pragma unroll
        for (int uh = 0; uh < 2; ++uh) {
            int u = uh * 16 + l16;
            #pragma unroll
            for (int m = 0; m < 2; ++m) {
                #pragma unroll
                for (int r = 0; r < 8; ++r) {
                    int cc = c0 + m * 16 + r + (hi ? 8 : 0);
                    float val = (u < VV) ? acc[tq][uh][m].f[r] + b3[cc] : 0.f;
                    x3p[(((size_t)n * COUT + cc) * TT + t) * U32 + u] = (__bf16)val;
                }
            }
        }
    }
}

// ---------------- K2: build attnT[n][o][v][u] (bf16), one block per n ----------------
__global__ __launch_bounds__(256) void ctrgc_attn_kernel(
    const float* __restrict__ xm_part, const float* __restrict__ Aadj,
    const float* __restrict__ w1, const float* __restrict__ b1,
    const float* __restrict__ w2, const float* __restrict__ b2,
    const float* __restrict__ w4, const float* __restrict__ b4,
    const int*  __restrict__ alpha_p,
    __bf16* __restrict__ attnT)
{
    __shared__ float xml[CIN * VV];          // 25.6 KB
    __shared__ float w1l[CR * CIN];          // 32 KB
    __shared__ float w2l[CR * CIN];          // 32 KB
    __shared__ float x1l[CR * VV];
    __shared__ float x2l[CR * VV];
    __shared__ float Al[VV * VV];
    __shared__ float b4l[COUT];
    __shared__ __attribute__((aligned(16))) __bf16 w4l[COUT * CR];   // 16 KB  [o][r]
    __shared__ __attribute__((aligned(16))) __bf16 afb[NPAD * CR];   // 40 KB  [uv][r]

    int n = blockIdx.x, tid = threadIdx.x;
    float alpha = (float)(*alpha_p);

    const float* p0 = xm_part + (size_t)n * CIN * VV;
    const float* p1 = xm_part + ((size_t)NN + n) * CIN * VV;
    for (int i = tid; i < CIN * VV; i += 256) xml[i] = (p0[i] + p1[i]) * (1.0f / TT);
    for (int i = tid; i < CR * CIN; i += 256) { w1l[i] = w1[i]; w2l[i] = w2[i]; }
    for (int i = tid; i < VV * VV; i += 256)  Al[i] = Aadj[i];
    for (int i = tid; i < COUT; i += 256)     b4l[i] = b4[i];
    for (int i = tid; i < COUT * CR; i += 256) w4l[i] = (__bf16)w4[i];
    __syncthreads();

    // x1, x2 (fp32)
    for (int i = tid; i < CR * VV; i += 256) {
        int r = i / VV, v = i % VV;
        float s1 = b1[r], s2 = b2[r];
        for (int c = 0; c < CIN; ++c) {
            float xv = xml[c * VV + v];
            s1 = fmaf(w1l[r * CIN + c], xv, s1);
            s2 = fmaf(w2l[r * CIN + c], xv, s2);
        }
        x1l[i] = s1; x2l[i] = s2;
    }
    __syncthreads();

    // aff = tanh(x1[u]-x2[v]) -> bf16, K(r)-contiguous layout for B-fragments
    for (int i = tid; i < VV * VV * CR; i += 256) {
        int r = i / (VV * VV); int rem = i % (VV * VV);
        int u = rem / VV, v = rem % VV;
        afb[(u * VV + v) * CR + r] = (__bf16)tanhf(x1l[r * VV + u] - x2l[r * VV + v]);
    }
    for (int i = tid; i < (NPAD - VV * VV) * CR; i += 256)
        afb[VV * VV * CR + i] = (__bf16)0.f;
    __syncthreads();

    // GEMM: attn[o][uv] = w4[o][r] * aff[r][uv], M=256 N=640 K=32
    int wave = tid >> 5, lane = tid & 31;
    int l16 = lane & 15; bool hi = lane >= 16;
    int ak = hi ? 8 : 0, bk = hi ? 8 : 0;
    for (int tile = wave; tile < 16 * 40; tile += 8) {
        int mt = tile / 40, nt = tile % 40;
        BF16x16 a, b;
        int arow = mt * 16 + l16;
        a.h[0] = *(const v8bf*)&w4l[arow * CR + ak];
        a.h[1] = *(const v8bf*)&w4l[arow * CR + ak + 16];
        int brow = nt * 16 + l16;
        b.h[0] = *(const v8bf*)&afb[brow * CR + bk];
        b.h[1] = *(const v8bf*)&afb[brow * CR + bk + 16];
        V8F acc = {};
        acc.v = __builtin_amdgcn_wmma_f32_16x16x32_bf16(
            false, a.v, false, b.v, (short)0, acc.v, false, false);
        int col = nt * 16 + l16;
        if (col < VV * VV) {
            int u = col / VV, v = col % VV;
            #pragma unroll
            for (int r = 0; r < 8; ++r) {
                int o = mt * 16 + r + (hi ? 8 : 0);
                float val = (acc.f[r] + b4l[o]) * alpha + Al[col];
                attnT[(((size_t)n * COUT + o) * U32 + v) * U32 + u] = (__bf16)val;
            }
        }
    }
    // zero the pad region (u>=25 or v>=25); disjoint from the GEMM's stores
    {
        __bf16 z = (__bf16)0.f;
        __bf16* base = attnT + ((size_t)n * COUT + tid) * U32 * U32;
        for (int v = 0; v < U32; ++v)
            for (int u = 0; u < U32; ++u)
                if (u >= VV || v >= VV) base[v * U32 + u] = z;
    }
}

// ---------------- K4: out[n][c][t][v] = sum_u x3[t][u]*attn[u][v], block per (n,c) ----------------
__global__ __launch_bounds__(128) void ctrgc_gemm2_kernel(
    const __bf16* __restrict__ x3p, const __bf16* __restrict__ attnT,
    float* __restrict__ out)
{
    int bid = blockIdx.x;                 // n*COUT + c
    int tid = threadIdx.x;
    int wave = tid >> 5, lane = tid & 31;
    int l16 = lane & 15; bool hi = lane >= 16;

    const __bf16* abase = x3p   + (size_t)bid * TT * U32;
    const __bf16* bbase = attnT + (size_t)bid * U32 * U32;
    float*        obase = out   + (size_t)bid * TT * VV;

    BF16x16 a;
    int ak = hi ? 8 : 0;
    int arow = wave * 16 + l16;           // t row
    a.h[0] = *(const v8bf*)&abase[arow * U32 + ak];
    a.h[1] = *(const v8bf*)&abase[arow * U32 + ak + 16];

    #pragma unroll
    for (int nt = 0; nt < 2; ++nt) {
        BF16x16 b;
        int brow = nt * 16 + l16;         // v column of attn (stored [v][u])
        int bk = hi ? 8 : 0;
        b.h[0] = *(const v8bf*)&bbase[brow * U32 + bk];
        b.h[1] = *(const v8bf*)&bbase[brow * U32 + bk + 16];
        V8F acc = {};
        acc.v = __builtin_amdgcn_wmma_f32_16x16x32_bf16(
            false, a.v, false, b.v, (short)0, acc.v, false, false);
        int v = nt * 16 + l16;
        if (v < VV) {
            #pragma unroll
            for (int r = 0; r < 8; ++r) {
                int t = wave * 16 + r + (hi ? 8 : 0);
                // write-once output: non-temporal store, keep L2 for intermediates
                __builtin_nontemporal_store(acc.f[r], obase + t * VV + v);
            }
        }
    }
}

extern "C" void kernel_launch(void* const* d_in, const int* in_sizes, int n_in,
                              void* d_out, int out_size, void* d_ws, size_t ws_size,
                              hipStream_t stream) {
    const float* x  = (const float*)d_in[0];
    const float* A  = (const float*)d_in[1];
    const float* w1 = (const float*)d_in[2];
    const float* b1 = (const float*)d_in[3];
    const float* w2 = (const float*)d_in[4];
    const float* b2 = (const float*)d_in[5];
    const float* w3 = (const float*)d_in[6];
    const float* b3 = (const float*)d_in[7];
    const float* w4 = (const float*)d_in[8];
    const float* b4 = (const float*)d_in[9];
    const int* alpha = (const int*)d_in[11];   // python scalar 1
    float* out = (float*)d_out;

    // workspace layout (256B-aligned offsets): total ~104 MB
    char* ws = (char*)d_ws;
    float*  xm_part = (float*)ws;                                  // 2 * 1,638,400 B
    __bf16* attnT   = (__bf16*)(ws + 2 * 1638400);                 // 33,554,432 B
    __bf16* x3p     = (__bf16*)(ws + 2 * 1638400 + 33554432);      // 67,108,864 B

    ctrgc_gemm1_mean_kernel<<<NN * 8 * 2, 512, 0, stream>>>(x, w3, b3, x3p, xm_part);
    ctrgc_attn_kernel      <<<NN,         256, 0, stream>>>(xm_part, A, w1, b1, w2, b2, w4, b4, alpha, attnT);
    ctrgc_gemm2_kernel     <<<NN * COUT,  128, 0, stream>>>(x3p, attnT, out);
}
